// SimpleNet_59957743452502
// MI455X (gfx1250) — matrix operand order repaired and checked by reading, not compiled, not run on hardware
//
#include <hip/hip_runtime.h>
#include <hip/hip_bf16.h>
#include <math.h>

// ---------------------------------------------------------------------------
// SoftKNN pipeline for MI455X (gfx1250, wave32, WMMA).
//   N=8192, D_in=512, H=256, MIN_K=20 -> 21st-smallest threshold per row.
// bf16 WMMA (f32 accumulate) for both N^2 GEMMs; the 8192^2 distance matrix
// lives only in registers/LDS tiles. All re-streamed operands (12 MB bf16)
// fit in the 192 MB L2, so HBM traffic is ~40 MB -> compute (WMMA) bound.
// k_out uses a 4-wave cooperative workgroup so the Gram GEMM is computed
// exactly once per i-tile (no per-d-chunk redundancy).
// ---------------------------------------------------------------------------

#define NPTS   8192
#define D_IN   512
#define HFEAT  256
#define KSEL   21      // MIN_K + 1 (self distance included, as in reference)
#define JS     8       // j-slices for the selection pass

typedef __bf16 bf16_t;
typedef __attribute__((ext_vector_type(16))) __bf16 v16bf;
typedef __attribute__((ext_vector_type(8)))  float  v8f;

// ---------------- WMMA helpers (layouts per CDNA5 ISA 7.12.2) --------------

__device__ __forceinline__ v8f wmma_bf16(v16bf a, v16bf b, v8f c) {
  // v_wmma_f32_16x16x32_bf16 : D = A(16x32) * B(32x16) + C(16x16 f32)
  return __builtin_amdgcn_wmma_f32_16x16x32_bf16(
      /*neg_a=*/false, a, /*neg_b=*/false, b,
      /*c_mod=*/(short)0, c, /*reuse_a=*/false, /*reuse_b=*/false);
}

// A-matrix 16x32 bf16: lane holds row m = lane%16; lanes<16 carry K {0..7,16..23},
// lanes>=16 carry K {8..15,24..31}. Source is row-major with leading dim ld.
__device__ __forceinline__ v16bf load_A_bf16(const bf16_t* base, int ld,
                                             int row0, int k0, int lane) {
  const int m  = row0 + (lane & 15);
  const int c0 = (lane < 16) ? 0 : 8;
  const bf16_t* p = base + (size_t)m * ld + k0;
  v16bf a;
#pragma unroll
  for (int t = 0; t < 8; ++t) a[t]     = p[c0 + t];
#pragma unroll
  for (int t = 0; t < 8; ++t) a[8 + t] = p[16 + c0 + t];
  return a;
}

// B-matrix 32x16 bf16 where B = srcRowsAlongK^T : column n of B is row
// (n0 + lane%16) of src; lanes<16 carry K {0..7,16..23}, lanes>=16 K {8..15,24..31}.
__device__ __forceinline__ v16bf load_B_bf16(const bf16_t* base, int ld,
                                             int n0, int k0, int lane) {
  const int n  = n0 + (lane & 15);
  const int kk = k0 + ((lane >> 4) << 3);
  const bf16_t* p = base + (size_t)n * ld + kk;
  v16bf b;
#pragma unroll
  for (int t = 0; t < 8; ++t) b[t] = p[t];
#pragma unroll
  for (int t = 0; t < 8; ++t) b[8 + t] = p[16 + t];
  return b;
}

// ---------------- 1a) x -> xT (bf16) and xb (bf16 row-major) ---------------

__global__ void k_transpose(const float* __restrict__ x, bf16_t* __restrict__ xT,
                            bf16_t* __restrict__ xb) {
  __shared__ float tile[16][17];
  const int i0 = blockIdx.x * 16;  // row of x
  const int d0 = blockIdx.y * 16;  // col of x
  const int tx = threadIdx.x, ty = threadIdx.y;
  const float v = x[(size_t)(i0 + ty) * D_IN + d0 + tx];
  tile[ty][tx] = v;
  xb[(size_t)(i0 + ty) * D_IN + d0 + tx] = (bf16_t)v;   // one-time bf16 copy
  __syncthreads();
  xT[(size_t)(d0 + ty) * NPTS + i0 + tx] = (bf16_t)tile[tx][ty];
}

// ---------------- 1b) W1 -> bf16 (one-time convert) ------------------------

__global__ void k_cvt_w1(const float* __restrict__ W1, bf16_t* __restrict__ W1b) {
  const int idx = blockIdx.x * 256 + threadIdx.x;
  W1b[idx] = (bf16_t)W1[idx];
}

// ---------------- 2) features = x @ W1^T  (bf16 out) -----------------------

__global__ void k_features(const bf16_t* __restrict__ xb, const bf16_t* __restrict__ W1b,
                           bf16_t* __restrict__ fb) {
  const int it   = blockIdx.x;          // i-tile (N/16)
  const int ht   = blockIdx.y;          // h-tile (H/16)
  const int lane = threadIdx.x;
  v8f c = {};
  for (int k0 = 0; k0 < D_IN; k0 += 32) {
    v16bf a = load_A_bf16(xb,  D_IN, it * 16, k0, lane);
    v16bf b = load_B_bf16(W1b, D_IN, ht * 16, k0, lane);
    c = wmma_bf16(a, b, c);
  }
  const int n  = ht * 16 + (lane & 15);
  const int mb = it * 16 + ((lane >> 4) << 3);
#pragma unroll
  for (int v = 0; v < 8; ++v)
    fb[(size_t)(mb + v) * HFEAT + n] = (bf16_t)c[v];
}

// ---------------- 3) sq[i] = ||f_i||^2  (from bf16 features) ---------------

__global__ void k_sq(const bf16_t* __restrict__ fb, float* __restrict__ sq) {
  const int row  = blockIdx.x * 8 + (threadIdx.x >> 5);
  const int lane = threadIdx.x & 31;
  const bf16_t* p = fb + (size_t)row * HFEAT;
  float s = 0.0f;
  for (int k = lane; k < HFEAT; k += 32) {
    float v = (float)p[k];
    s += v * v;
  }
#pragma unroll
  for (int off = 16; off > 0; off >>= 1) s += __shfl_xor(s, off, 32);
  if (lane == 0) sq[row] = s;
}

// ---------------- 4) partial top-21 distances per (i-tile, j-slice) --------

__global__ void k_select(const bf16_t* __restrict__ fb, const float* __restrict__ sq,
                         float* __restrict__ cand) {
  const int it   = blockIdx.x;
  const int js   = blockIdx.y;
  const int lane = threadIdx.x;
  const int i0   = it * 16;
  const int mb   = (lane >> 4) << 3;
  __shared__ float dt[16][17];

  float sqm[8];
#pragma unroll
  for (int v = 0; v < 8; ++v) sqm[v] = sq[i0 + mb + v];

  float a[KSEL];
#pragma unroll
  for (int t = 0; t < KSEL; ++t) a[t] = 3.0e38f;

  // A operand (this i-tile's features) is reused for every j-tile: keep in regs.
  v16bf areg[8];
#pragma unroll
  for (int kk = 0; kk < 8; ++kk)
    areg[kk] = load_A_bf16(fb, HFEAT, i0, kk * 32, lane);

  const int jbeg   = js * (NPTS / JS);
  const int jtiles = (NPTS / JS) / 16;
  for (int jt = 0; jt < jtiles; ++jt) {
    const int j0 = jbeg + jt * 16;
    if (jt + 1 < jtiles)
      __builtin_prefetch(fb + (size_t)(j0 + 16 + (lane & 15)) * HFEAT, 0, 0);
    v8f g = {};
#pragma unroll
    for (int kk = 0; kk < 8; ++kk) {
      v16bf b = load_B_bf16(fb, HFEAT, j0, kk * 32, lane);
      g = wmma_bf16(areg[kk], b, g);
    }
    const float sqn = sq[j0 + (lane & 15)];
#pragma unroll
    for (int v = 0; v < 8; ++v) {
      float d2 = sqm[v] + sqn - 2.0f * g[v];
      float d  = sqrtf(fmaxf(d2, 1e-12f));
      dt[mb + v][lane & 15] = d;          // C layout -> row-major LDS tile
    }
    asm volatile("s_wait_dscnt 0x0" ::: "memory");  // intra-wave LDS transpose
    if (lane < 16) {                      // lane owns row (i0 + lane)
#pragma unroll
      for (int q = 0; q < 16; ++q) {
        float xv = dt[lane][q];
        if (xv < a[KSEL - 1]) {           // early reject keeps VALU cost low
          a[KSEL - 1] = xv;
#pragma unroll
          for (int t = KSEL - 1; t > 0; --t) {
            float lo = fminf(a[t - 1], a[t]);
            float hi = fmaxf(a[t - 1], a[t]);
            a[t - 1] = lo; a[t] = hi;
          }
        }
      }
    }
  }
  if (lane < 16) {
    float* c = cand + ((size_t)(i0 + lane) * JS + js) * KSEL;
#pragma unroll
    for (int t = 0; t < KSEL; ++t) c[t] = a[t];
  }
}

// ---------------- 5) merge slices -> kth distance per row ------------------

__global__ void k_kth(const float* __restrict__ cand, float* __restrict__ kth) {
  const int row = blockIdx.x * blockDim.x + threadIdx.x;
  if (row >= NPTS) return;
  const float* c = cand + (size_t)row * (JS * KSEL);
  float a[KSEL];
#pragma unroll
  for (int t = 0; t < KSEL; ++t) a[t] = 3.0e38f;
  for (int q = 0; q < JS * KSEL; ++q) {
    float xv = c[q];
    if (xv < a[KSEL - 1]) {
      a[KSEL - 1] = xv;
#pragma unroll
      for (int t = KSEL - 1; t > 0; --t) {
        float lo = fminf(a[t - 1], a[t]);
        float hi = fmaxf(a[t - 1], a[t]);
        a[t - 1] = lo; a[t] = hi;
      }
    }
  }
  kth[row] = a[KSEL - 1];
}

// ---------------- 6) out = rownorm(relu(kth - d)) @ x ----------------------
// 4-wave cooperative workgroup per 16-row i-tile:
//   phase 1: wave w computes Gram/weights for j-subtile (j0 + 16w), so the
//            Gram GEMM is computed exactly once per i-tile across the block;
//            w-tile (16x64 f32) shared through padded LDS.
//   phase 2: every wave consumes the full K=64 w-chunk (2 WMMA A operands)
//            against its own 128 output columns of x^T.
// Row sums accumulated in f32 and combined across waves at the end.

__global__ void __launch_bounds__(128) k_out(
    const bf16_t* __restrict__ fb, const bf16_t* __restrict__ xT,
    const float* __restrict__ sq, const float* __restrict__ kth,
    float* __restrict__ out) {
  const int it   = blockIdx.x;
  const int wave = threadIdx.x >> 5;
  const int lane = threadIdx.x & 31;
  const int i0   = it * 16;
  const int mb   = (lane >> 4) << 3;
  const int dblk = wave * 128;
  __shared__ float wt[16][65];     // padded: row stride 65 avoids bank conflicts
  __shared__ float rsum[4][16];

  float sqm[8], thr[8], rs[8];
#pragma unroll
  for (int v = 0; v < 8; ++v) {
    sqm[v] = sq[i0 + mb + v];
    thr[v] = kth[i0 + mb + v];
    rs[v]  = 0.0f;
  }

  v16bf areg[8];
#pragma unroll
  for (int kk = 0; kk < 8; ++kk)
    areg[kk] = load_A_bf16(fb, HFEAT, i0, kk * 32, lane);

  v8f acc[8] = {};  // 16 x 128 f32 accumulator (8 n-tiles) per wave

  for (int j0 = 0; j0 < NPTS; j0 += 64) {
    // ---- phase 1: this wave's 16-column j-subtile of the Gram/weight tile
    const int jb = j0 + wave * 16;
    if (j0 + 64 < NPTS)
      __builtin_prefetch(fb + (size_t)(j0 + 64 + wave * 16 + (lane & 15)) * HFEAT, 0, 0);
    v8f g = {};
#pragma unroll
    for (int kk = 0; kk < 8; ++kk) {
      v16bf b = load_B_bf16(fb, HFEAT, jb, kk * 32, lane);
      g = wmma_bf16(areg[kk], b, g);
    }
    const float sqn = sq[jb + (lane & 15)];
#pragma unroll
    for (int v = 0; v < 8; ++v) {
      float d2 = sqm[v] + sqn - 2.0f * g[v];
      float d  = sqrtf(fmaxf(d2, 1e-12f));
      float w  = fmaxf(thr[v] - d, 0.0f);
      rs[v] += w;
      wt[mb + v][wave * 16 + (lane & 15)] = w;
    }
    __syncthreads();

    // ---- phase 2: consume full K=64 chunk against this wave's 128 columns
    const int m  = lane & 15;
    const int c0 = (lane < 16) ? 0 : 8;
#pragma unroll
    for (int c = 0; c < 2; ++c) {
      v16bf aw;
#pragma unroll
      for (int t = 0; t < 8; ++t) aw[t]     = (bf16_t)wt[m][c * 32 + c0 + t];
#pragma unroll
      for (int t = 0; t < 8; ++t) aw[8 + t] = (bf16_t)wt[m][c * 32 + 16 + c0 + t];
#pragma unroll
      for (int nt = 0; nt < 8; ++nt) {
        v16bf b = load_B_bf16(xT, NPTS, dblk + nt * 16, j0 + c * 32, lane);
        acc[nt] = wmma_bf16(aw, b, acc[nt]);
      }
    }
    __syncthreads();   // protect wt before next iteration overwrites it
  }

  // Row sums: reduce across the 16 lanes of each half-wave (same C-layout row),
  // then combine the 4 per-wave partials (each wave covered 1/4 of j).
#pragma unroll
  for (int v = 0; v < 8; ++v) {
    float s = rs[v];
    s += __shfl_xor(s, 1, 16);
    s += __shfl_xor(s, 2, 16);
    s += __shfl_xor(s, 4, 16);
    s += __shfl_xor(s, 8, 16);
    rs[v] = s;
  }
  if ((lane & 15) == 0) {
#pragma unroll
    for (int v = 0; v < 8; ++v) rsum[wave][mb + v] = rs[v];
  }
  __syncthreads();

#pragma unroll
  for (int v = 0; v < 8; ++v) {
    const float tot = rsum[0][mb + v] + rsum[1][mb + v] +
                      rsum[2][mb + v] + rsum[3][mb + v];
    const float inv = 1.0f / tot;
    const size_t rowoff = (size_t)(i0 + mb + v) * D_IN;
#pragma unroll
    for (int nt = 0; nt < 8; ++nt)
      out[rowoff + dblk + nt * 16 + (lane & 15)] = acc[nt][v] * inv;
  }
}

// ---------------------------------------------------------------------------

extern "C" void kernel_launch(void* const* d_in, const int* in_sizes, int n_in,
                              void* d_out, int out_size, void* d_ws, size_t ws_size,
                              hipStream_t stream) {
  (void)in_sizes; (void)n_in; (void)out_size; (void)ws_size;
  const float* x  = (const float*)d_in[0];
  // d_in[1] is y: unused by the reference computation.
  const float* W1 = (const float*)d_in[2];
  float* out = (float*)d_out;

  // Workspace layout (~26.5 MB total).
  char* ws = (char*)d_ws;
  bf16_t* xT   = (bf16_t*)ws;  ws += (size_t)D_IN * NPTS * sizeof(bf16_t);   // 8 MB
  bf16_t* xb   = (bf16_t*)ws;  ws += (size_t)NPTS * D_IN * sizeof(bf16_t);   // 8 MB
  bf16_t* fb   = (bf16_t*)ws;  ws += (size_t)NPTS * HFEAT * sizeof(bf16_t);  // 4 MB
  bf16_t* W1b  = (bf16_t*)ws;  ws += (size_t)HFEAT * D_IN * sizeof(bf16_t);  // 256 KB
  float*  sq   = (float*)ws;   ws += (size_t)NPTS * sizeof(float);
  float*  kth  = (float*)ws;   ws += (size_t)NPTS * sizeof(float);
  float*  cand = (float*)ws;   ws += (size_t)NPTS * JS * KSEL * sizeof(float);

  k_transpose<<<dim3(NPTS / 16, D_IN / 16), dim3(16, 16), 0, stream>>>(x, xT, xb);
  k_cvt_w1<<<(HFEAT * D_IN) / 256, 256, 0, stream>>>(W1, W1b);
  k_features<<<dim3(NPTS / 16, HFEAT / 16), 32, 0, stream>>>(xb, W1b, fb);
  k_sq<<<NPTS / 8, 256, 0, stream>>>(fb, sq);
  k_select<<<dim3(NPTS / 16, JS), 32, 0, stream>>>(fb, sq, cand);
  k_kth<<<NPTS / 256, 256, 0, stream>>>(cand, kth);
  k_out<<<NPTS / 16, 128, 0, stream>>>(fb, xT, sq, kth, out);
}
